// CondensationDiagnostics_24515673325994
// MI455X (gfx1250) — compile-verified, matrix operand order checked by reading, not run on hardware
//
#include <hip/hip_runtime.h>
#include <hip/hip_bf16.h>

typedef _Float16 v16h __attribute__((ext_vector_type(16)));
typedef _Float16 v8h  __attribute__((ext_vector_type(8)));
typedef float    v8f  __attribute__((ext_vector_type(8)));

#define N_AGENTS 4096
#define M_CLUST  128
#define K_DIM    32
#define G_SP     16
#define W_EPS    1e-6f

// ---------------------------------------------------------------------------
// Kernel 1: W (N, M, G) --mean over G--> WT (M, N)  (transposed via LDS tile)
// ---------------------------------------------------------------------------
__global__ __launch_bounds__(256) void k_wmean_t(const float* __restrict__ W,
                                                 float* __restrict__ WT) {
  __shared__ float tile[32][33];
  const int i0 = blockIdx.x * 32;
  const int a0 = blockIdx.y * 32;
  #pragma unroll
  for (int p = 0; p < 4; ++p) {
    int idx = p * 256 + threadIdx.x;        // 0..1023
    int ai = idx & 31, ii = idx >> 5;
    const float* src = W + ((size_t)(i0 + ii) * M_CLUST + (a0 + ai)) * G_SP;
    float s = 0.f;
    #pragma unroll
    for (int g = 0; g < G_SP; ++g) s += src[g];
    tile[ii][ai] = s * (1.f / G_SP);
  }
  __syncthreads();
  #pragma unroll
  for (int p = 0; p < 4; ++p) {
    int idx = p * 256 + threadIdx.x;
    int wi = idx & 31, wa = idx >> 5;
    WT[(size_t)(a0 + wa) * N_AGENTS + i0 + wi] = tile[wi][wa];
  }
}

// ---------------------------------------------------------------------------
// Kernel 2: omega_parent (M, G, K, K) --mean over G--> OpH (M, K, K) in f16
// ---------------------------------------------------------------------------
__global__ __launch_bounds__(256) void k_opmean_h(const float* __restrict__ OP,
                                                  _Float16* __restrict__ OpH) {
  int o = blockIdx.x * 256 + threadIdx.x;   // < M*K*K = 131072
  int m = o >> 10, e = o & 1023;
  const float* src = OP + (size_t)m * G_SP * 1024 + e;
  float s = 0.f;
  #pragma unroll
  for (int g = 0; g < G_SP; ++g) s += src[(size_t)g * 1024];
  OpH[o] = (_Float16)(s * (1.f / G_SP));
}

// ---------------------------------------------------------------------------
// Kernel 3: fused omega_child G-average + mu G-average + SPD solve
//   v[i] = inv(mean_g oc[i,g]) @ mean_g mu[i,g]   -> Vh (N, K) f16
// One wave per matrix (lane = row), Gauss-Jordan in LDS (SPD: no pivoting).
// This kernel owns the dominant 256 MB HBM stream.
// ---------------------------------------------------------------------------
__global__ __launch_bounds__(256) void k_solve(const float* __restrict__ OC,
                                               const float* __restrict__ MU,
                                               _Float16* __restrict__ Vh) {
  __shared__ float A[8][32][34];            // 8 waves x (32x33 augmented, +pad)
  const int wave = threadIdx.x >> 5;
  const int r    = threadIdx.x & 31;
  const int i    = blockIdx.x * 8 + wave;

  // average omega_child[i, :, r, :] over G into registers (coalesced per g)
  const float* base = OC + (size_t)i * G_SP * 1024 + (size_t)r * 32;
  float acc[32];
  #pragma unroll
  for (int c = 0; c < 32; ++c) acc[c] = 0.f;
  for (int g = 0; g < G_SP; ++g) {
    const float* row = base + (size_t)g * 1024;
    #pragma unroll
    for (int c = 0; c < 32; ++c) acc[c] += row[c];
  }
  #pragma unroll
  for (int c = 0; c < 32; ++c) A[wave][r][c] = acc[c] * (1.f / G_SP);

  float rs = 0.f;
  #pragma unroll
  for (int g = 0; g < G_SP; ++g) rs += MU[((size_t)i * G_SP + g) * K_DIM + r];
  A[wave][r][32] = rs * (1.f / G_SP);
  __syncthreads();

  // Gauss-Jordan: after 32 steps the matrix is diagonal.
  for (int p = 0; p < 32; ++p) {
    float piv = A[wave][p][p];
    float f = (r == p) ? 0.f : A[wave][r][p] / piv;
    #pragma unroll
    for (int c = 0; c < 33; ++c) A[wave][r][c] -= f * A[wave][p][c];
    __syncthreads();
  }
  float v = A[wave][r][32] / A[wave][r][r];
  Vh[(size_t)i * K_DIM + r] = (_Float16)v;
}

// ---------------------------------------------------------------------------
// Kernel 4: per-cluster WMMA stats.
//   Y = Vh @ OpH[a]^T via v_wmma_f32_16x16x32_f16 (contraction = K = 32),
//   accumulate sum(w*y) and sum(w*y^2) per k in registers; psi via
//   E[y^2] - E[y]^2 identity. One workgroup (8 waves) per cluster a.
// ---------------------------------------------------------------------------
__global__ __launch_bounds__(256) void k_stats(const _Float16* __restrict__ Vh,
                                               const _Float16* __restrict__ OpH,
                                               const float* __restrict__ WT,
                                               float* __restrict__ out) {
  __shared__ float s1[8][32], s2[8][32], sz[8], sc[8];
  const int a    = blockIdx.x;
  const int tid  = threadIdx.x;
  const int wave = tid >> 5;
  const int lane = tid & 31;
  const int sel  = lane >> 4;     // which half of the 16-bit fragment layout
  const int lr   = lane & 15;     // row within tile

  // 16-bit fragment loader per ISA layout:
  //   lanes 0-15 : K 0-7 then 16-23 ; lanes 16-31 : K 8-15 then 24-31
  auto load_frag = [&](const _Float16* row) -> v16h {
    v8h lo = *(const v8h*)(row + sel * 8);
    v8h hi = *(const v8h*)(row + 16 + sel * 8);
    v16h f;
    #pragma unroll
    for (int e = 0; e < 8; ++e) { f[e] = lo[e]; f[8 + e] = hi[e]; }
    return f;
  };

  // B fragments: rows k of OpH[a] (row-major => B used as Op^T, same layout as A)
  const _Float16* opa = OpH + (size_t)a * 1024;
  v16h b0 = load_frag(opa + (size_t)lr * 32);
  v16h b1 = load_frag(opa + (size_t)(16 + lr) * 32);

  float acc1a = 0.f, acc2a = 0.f, acc1b = 0.f, acc2b = 0.f;
  float zp = 0.f, cp = 0.f;

  for (int t = wave; t < N_AGENTS / 16; t += 8) {   // wave-uniform loop
    v16h af = load_frag(Vh + (size_t)(t * 16 + lr) * K_DIM);
    float wv = (lane < 16) ? WT[(size_t)a * N_AGENTS + t * 16 + lane] : 0.f;
    float we = (wv >= W_EPS) ? wv : 0.f;
    zp += we;
    cp += (lane < 16 && wv >= W_EPS) ? 1.f : 0.f;

    v8f c0 = {0.f, 0.f, 0.f, 0.f, 0.f, 0.f, 0.f, 0.f};
    v8f d0 = __builtin_amdgcn_wmma_f32_16x16x32_f16(false, af, false, b0,
                                                    (short)0, c0, false, false);
    v8f d1 = __builtin_amdgcn_wmma_f32_16x16x32_f16(false, af, false, b1,
                                                    (short)0, c0, false, false);
    // D layout: d[j] = Y[i = t*16 + j + 8*sel, k = ktile*16 + lr]
    #pragma unroll
    for (int j = 0; j < 8; ++j) {
      float wj = __shfl(we, j + sel * 8, 32);
      float y0 = d0[j], y1 = d1[j];
      acc1a += wj * y0;  acc2a += wj * y0 * y0;
      acc1b += wj * y1;  acc2b += wj * y1 * y1;
    }
  }

  // lanes L and L^16 hold the same k: fold, then stage per-wave partials in LDS
  acc1a += __shfl_xor(acc1a, 16, 32);
  acc2a += __shfl_xor(acc2a, 16, 32);
  acc1b += __shfl_xor(acc1b, 16, 32);
  acc2b += __shfl_xor(acc2b, 16, 32);
  #pragma unroll
  for (int off = 16; off >= 1; off >>= 1) {
    zp += __shfl_xor(zp, off, 32);
    cp += __shfl_xor(cp, off, 32);
  }
  if (lane < 16) {
    s1[wave][lane]      = acc1a;  s1[wave][16 + lane] = acc1b;
    s2[wave][lane]      = acc2a;  s2[wave][16 + lane] = acc2b;
  }
  if (lane == 0) { sz[wave] = zp; sc[wave] = cp; }
  __syncthreads();

  if (tid < 32) {   // wave 0 finishes: deterministic fixed-order sums
    float t1 = 0.f, t2 = 0.f, Z = 0.f, C = 0.f;
    #pragma unroll
    for (int w2 = 0; w2 < 8; ++w2) {
      t1 += s1[w2][tid];  t2 += s2[w2][tid];
      Z  += sz[w2];       C  += sc[w2];
    }
    float Zs = (Z > 0.f) ? Z : 1.f;
    float m = t1 / Zs;
    float term = t2 / Zs - m * m;   // var contribution for k = tid
    #pragma unroll
    for (int off = 16; off >= 1; off >>= 1) term += __shfl_xor(term, off, 32);
    if (tid == 0) out[a] = (C >= 1.5f) ? term : 0.f;
  }
}

// ---------------------------------------------------------------------------
extern "C" void kernel_launch(void* const* d_in, const int* in_sizes, int n_in,
                              void* d_out, int out_size, void* d_ws, size_t ws_size,
                              hipStream_t stream) {
  const float* W  = (const float*)d_in[0];   // (N, M, G)
  const float* MU = (const float*)d_in[1];   // (N, G, K)
  const float* OC = (const float*)d_in[2];   // (N, G, K, K)
  const float* OP = (const float*)d_in[3];   // (M, G, K, K)
  float* out = (float*)d_out;                // (M,)

  char* ws = (char*)d_ws;
  float*    WT  = (float*)ws;                                        // M*N f32 (2 MiB)
  _Float16* Vh  = (_Float16*)(ws + (size_t)M_CLUST * N_AGENTS * 4);  // N*K f16
  _Float16* OpH = (_Float16*)(ws + (size_t)M_CLUST * N_AGENTS * 4
                                 + (size_t)N_AGENTS * K_DIM * 2);    // M*K*K f16

  k_wmean_t <<<dim3(N_AGENTS / 32, M_CLUST / 32), 256, 0, stream>>>(W, WT);
  k_opmean_h<<<dim3((M_CLUST * K_DIM * K_DIM) / 256), 256, 0, stream>>>(OP, OpH);
  k_solve   <<<dim3(N_AGENTS / 8), 256, 0, stream>>>(OC, MU, Vh);
  k_stats   <<<dim3(M_CLUST), 256, 0, stream>>>(Vh, OpH, WT, out);
}
